// PureCartesianE3Conv2_19172734009892
// MI455X (gfx1250) — matrix operand order repaired and checked by reading, not compiled, not run on hardware
//
#include <hip/hip_runtime.h>
#include <hip/hip_bf16.h>
#include <math.h>

// ---------------- problem constants ----------------
#define N_NODES   4096
#define N_EDGES   49152
#define CCH       64
#define NB        16
#define NPATH     11
#define WNUMEL    (NPATH * CCH)      // 704
#define DIM_IN    (CCH * 26)         // 1664
#define MAXR      4.0f
#define WAVES     4                  // waves per block (wave32)
#define TILES     (N_EDGES / 16)     // 3072 tiles of 16 edges
#define BLOCKS    (TILES / WAVES)    // 768
#define NFRAG     100                // 4 (W1) + 8 (W2) + 88 (W3) B-fragments
#define FRAG_ELEMS 512               // 32 lanes x 16 f16 per fragment

typedef __attribute__((ext_vector_type(16))) _Float16 v16h;
typedef __attribute__((ext_vector_type(8)))  float    v8f;

#define WMMA(A, B, C) __builtin_amdgcn_wmma_f32_16x16x32_f16(false, (A), false, (B), (short)0, (C), false, false)

// ---------- fragment helpers (CDNA5 WMMA 16x16x32 f16, wave32) ----------

// Packed B fragment: lane reads 16 contiguous f16 (32B -> 2x global_load_b128).
__device__ inline v16h load_bfrag_pk(const _Float16* __restrict__ pk, int frag) {
  int lane = threadIdx.x & 31;
  return *(const v16h*)(pk + (size_t)frag * FRAG_ELEMS + lane * 16);
}

// A fragment from LDS 16x64 f16 buffer, K-window [kbase, kbase+32).
// ISA layout: reg i<4 -> K=2i+kh, reg i>=4 -> K=16+2(i-4)+kh, kh = 8*(lane>=16).
__device__ inline v16h load_afrag(const _Float16* __restrict__ buf, int kbase) {
  int lane = threadIdx.x & 31;
  int m  = lane & 15;
  int kh = (lane >> 4) * 8;
  const _Float16* row = buf + m * 64 + kbase;
  v16h a;
#pragma unroll
  for (int i = 0; i < 4; ++i) {
    a[2 * i]         = row[kh + 2 * i];
    a[2 * i + 1]     = row[kh + 2 * i + 1];
    a[8 + 2 * i]     = row[16 + kh + 2 * i];
    a[8 + 2 * i + 1] = row[16 + kh + 2 * i + 1];
  }
  return a;
}

// C fragment broadcasting a bias row: C(m,n) = bias[colbase+n].
__device__ inline v8f bias_frag(const float* __restrict__ bias, int colbase) {
  float v = bias[colbase + (threadIdx.x & 15)];
  v8f c;
#pragma unroll
  for (int i = 0; i < 8; ++i) c[i] = v;
  return c;
}

__device__ inline float fast_silu(float x) {
  return x * __builtin_amdgcn_rcpf(1.0f + __expf(-x));
}

// D fragment (m,n): lane n (+16 if m>=8), vgpr m%8. Store silu(D) to f16 LDS 16x64.
__device__ inline void store_h_silu(_Float16* __restrict__ buf, int colbase, v8f d) {
  int lane = threadIdx.x & 31;
  int n  = lane & 15;
  int mb = (lane >> 4) * 8;
#pragma unroll
  for (int r = 0; r < 8; ++r)
    buf[(mb + r) * 64 + colbase + n] = (_Float16)fast_silu(d[r]);
}

// Store raw f32 D fragment into LDS 16x16 tile.
__device__ inline void store_wtile(float* __restrict__ wb, v8f d) {
  int lane = threadIdx.x & 31;
  int n  = lane & 15;
  int mb = (lane >> 4) * 8;
#pragma unroll
  for (int r = 0; r < 8; ++r) wb[(mb + r) * 16 + n] = d[r];
}

__device__ inline float gauss_basis(float r, int k) {
  if (k >= NB) return 0.0f;
  float v = (MAXR / 17.0f) * (float)(k + 1);
  float d = (r - v) * (17.0f / MAXR);
  return __expf(-d * d) * (4.0f / 1.12f);   // sqrt(NB)=4, /1.12
}

// ---------------- weight pack kernel (one wave per fragment) ----------------
// Fragment layout: value(lane, i) = W[kbase + 8*(lane>=16) + i + 8*(i>=8)][colbase + (lane&15)]
__global__ __launch_bounds__(32)
void e3conv_pack(const float* __restrict__ W1, const float* __restrict__ W2,
                 const float* __restrict__ W3, _Float16* __restrict__ pk) {
  int frag = blockIdx.x;
  int lane = threadIdx.x;
  const float* W; int ldw, colbase, kbase, kmax;
  if (frag < 4)        { W = W1; ldw = 64;     colbase = frag * 16;        kbase = 0;            kmax = NB; }
  else if (frag < 12)  { int f = frag - 4;  W = W2; ldw = 64;     colbase = (f >> 1) * 16; kbase = (f & 1) * 32; kmax = 64; }
  else                 { int f = frag - 12; W = W3; ldw = WNUMEL; colbase = (f >> 1) * 16; kbase = (f & 1) * 32; kmax = 64; }
  int col = colbase + (lane & 15);
  int k0  = kbase + ((lane >> 4) << 3);
  _Float16* dst = pk + (size_t)frag * FRAG_ELEMS + lane * 16;
#pragma unroll
  for (int i = 0; i < 16; ++i) {
    int k = k0 + i + ((i >> 3) << 3);
    dst[i] = (k < kmax) ? (_Float16)W[(size_t)k * ldw + col] : (_Float16)0.0f;
  }
}

// ---------------- main fused kernel ----------------
__global__ __launch_bounds__(WAVES * 32)
void e3conv_main(const float* __restrict__ f_in, const float* __restrict__ pos,
                 const int* __restrict__ batch, const int* __restrict__ esrc,
                 const int* __restrict__ edst, const float* __restrict__ eshift,
                 const float* __restrict__ cell,
                 const float* __restrict__ b1, const float* __restrict__ b2,
                 const float* __restrict__ b3,
                 const _Float16* __restrict__ pk,
                 float* __restrict__ out, float* __restrict__ cnt) {
  __shared__ _Float16 h1buf[WAVES][16][64];
  __shared__ _Float16 h2buf[WAVES][16][64];
  __shared__ float    ubuf [WAVES][16][3];
  __shared__ float    rbuf [WAVES][16];
  __shared__ int      srcbuf[WAVES][16];
  __shared__ int      dstbuf[WAVES][16];
  __shared__ float    wbuf [WAVES][NPATH][16][16];   // weight tiles for one channel-group

  const int wave = threadIdx.x >> 5;
  const int lane = threadIdx.x & 31;
  const int tile = blockIdx.x * WAVES + wave;        // < TILES by construction

  // ---- geometry: lanes 0..15 handle one edge each ----
  if (lane < 16) {
    int e = tile * 16 + lane;
    int s = esrc[e], d = edst[e];
    int bg = batch[s];
    const float* ce = cell + bg * 9;
    float sh[3], ev[3];
#pragma unroll
    for (int j = 0; j < 3; ++j)
      sh[j] = eshift[e * 3 + 0] * ce[0 * 3 + j] +
              eshift[e * 3 + 1] * ce[1 * 3 + j] +
              eshift[e * 3 + 2] * ce[2 * 3 + j];
#pragma unroll
    for (int j = 0; j < 3; ++j)
      ev[j] = pos[d * 3 + j] - pos[s * 3 + j] + sh[j];
    float r = sqrtf(ev[0] * ev[0] + ev[1] * ev[1] + ev[2] * ev[2]);
    float inv = __builtin_amdgcn_rcpf(fmaxf(r, 1e-12f));
    rbuf[wave][lane] = r;
#pragma unroll
    for (int j = 0; j < 3; ++j) ubuf[wave][lane][j] = ev[j] * inv;
    srcbuf[wave][lane] = s;
    dstbuf[wave][lane] = d;
    atomicAdd(cnt + d, 1.0f);
  }
  __syncthreads();

  // ---- radial-embedding A fragment built in-register (K=16 real, padded to 32) ----
  {
    int m  = lane & 15;
    int kh = (lane >> 4) * 8;
    float re = rbuf[wave][m];
    v16h aemb;
#pragma unroll
    for (int i = 0; i < 8; ++i) {
      int k = (i < 4) ? (2 * i + kh) : (16 + 2 * (i - 4) + kh);
      aemb[2 * i]     = (_Float16)gauss_basis(re, k);
      aemb[2 * i + 1] = (_Float16)gauss_basis(re, k + 1);
    }
    // ---- GEMM1: h1 = silu(emb @ W1 + b1), 16x64 ----
#pragma unroll
    for (int j = 0; j < 4; ++j) {
      v16h bf = load_bfrag_pk(pk, j);
      v8f acc = bias_frag(b1, j * 16);
      acc = WMMA(aemb, bf, acc);
      store_h_silu(&h1buf[wave][0][0], j * 16, acc);
    }
  }
  __syncthreads();

  // ---- GEMM2: h2 = silu(h1 @ W2 + b2), K=64 ----
  {
    v16h a0 = load_afrag(&h1buf[wave][0][0], 0);
    v16h a1 = load_afrag(&h1buf[wave][0][0], 32);
#pragma unroll
    for (int j = 0; j < 4; ++j) {
      v8f acc = bias_frag(b2, j * 16);
      acc = WMMA(a0, load_bfrag_pk(pk, 4 + j * 2 + 0), acc);
      acc = WMMA(a1, load_bfrag_pk(pk, 4 + j * 2 + 1), acc);
      store_h_silu(&h2buf[wave][0][0], j * 16, acc);
    }
  }
  __syncthreads();

  // ---- GEMM3 (per channel-group) + tensor product + scatter ----
  v16h g0 = load_afrag(&h2buf[wave][0][0], 0);
  v16h g1 = load_afrag(&h2buf[wave][0][0], 32);

  for (int cg = 0; cg < 4; ++cg) {
    for (int p = 0; p < NPATH; ++p) {
      int ntile = p * 4 + cg;                 // column tile (16 cols each)
      v8f acc = bias_frag(b3, ntile * 16);
      acc = WMMA(g0, load_bfrag_pk(pk, 12 + ntile * 2 + 0), acc);
      acc = WMMA(g1, load_bfrag_pk(pk, 12 + ntile * 2 + 1), acc);
      store_wtile(&wbuf[wave][p][0][0], acc);
    }
    __syncthreads();

    // 32 lanes = 2 parities x 16 channels of this group; loop edges.
    const int par = lane >> 4;
    const int cl  = lane & 15;
    const int c   = cg * 16 + cl;
    const int off0 = par * 64 + c;
    const int off1 = 128 + par * 192 + c * 3;
    const int off2 = 512 + par * 576 + c * 9;

    for (int e = 0; e < 16; ++e) {
      float w[NPATH];
#pragma unroll
      for (int i = 0; i < NPATH; ++i) w[i] = wbuf[wave][i][e][cl];
      float uu[3];
#pragma unroll
      for (int j = 0; j < 3; ++j) uu[j] = ubuf[wave][e][j];

      const float* xrow = f_in + (size_t)srcbuf[wave][e] * DIM_IN;
      float*       orow = out  + (size_t)dstbuf[wave][e] * DIM_IN;

      float x0 = xrow[off0];
      float x1[3], x2[9];
#pragma unroll
      for (int j = 0; j < 3; ++j) x1[j] = xrow[off1 + j];
#pragma unroll
      for (int j = 0; j < 9; ++j) x2[j] = xrow[off2 + j];

      float d1 = x1[0] * uu[0] + x1[1] * uu[1] + x1[2] * uu[2];
      float v2[3];
#pragma unroll
      for (int a = 0; a < 3; ++a)
        v2[a] = x2[a * 3 + 0] * uu[0] + x2[a * 3 + 1] * uu[1] + x2[a * 3 + 2] * uu[2];
      float q2 = v2[0] * uu[0] + v2[1] * uu[1] + v2[2] * uu[2];

      // L=0 output
      atomicAdd(orow + off0, w[0] * x0 + w[5] * d1 + w[10] * q2);

      // L=1 output: (w1*x0 + w6*d1)*u + w3*x1 + w8*v2
      float t1 = w[1] * x0 + w[6] * d1;
#pragma unroll
      for (int b = 0; b < 3; ++b)
        atomicAdd(orow + off1 + b, t1 * uu[b] + w[3] * x1[b] + w[8] * v2[b]);

      // L=2 output: s[a]*u[b] + w7*x2[ab], s[a] = w2*x0*u[a] + w4*x1[a] + w9*v2[a]
#pragma unroll
      for (int a = 0; a < 3; ++a) {
        float s = w[2] * x0 * uu[a] + w[4] * x1[a] + w[9] * v2[a];
#pragma unroll
        for (int b = 0; b < 3; ++b)
          atomicAdd(orow + off2 + a * 3 + b, s * uu[b] + w[7] * x2[a * 3 + b]);
      }
    }
    __syncthreads();
  }
}

// ---------------- zero + normalize ----------------
__global__ void e3conv_zero(float* __restrict__ out, float* __restrict__ cnt, int n) {
  int i = blockIdx.x * blockDim.x + threadIdx.x;
  for (; i < n; i += gridDim.x * blockDim.x) out[i] = 0.0f;
  int j = blockIdx.x * blockDim.x + threadIdx.x;
  if (j < N_NODES) cnt[j] = 0.0f;
}

__global__ void e3conv_norm(float* __restrict__ out, const float* __restrict__ cnt, int n) {
  int i = blockIdx.x * blockDim.x + threadIdx.x;
  for (; i < n; i += gridDim.x * blockDim.x) {
    float c = fmaxf(cnt[i / DIM_IN], 1.0f);
    out[i] = out[i] / c;
  }
}

extern "C" void kernel_launch(void* const* d_in, const int* in_sizes, int n_in,
                              void* d_out, int out_size, void* d_ws, size_t ws_size,
                              hipStream_t stream) {
  const float* f_in   = (const float*)d_in[0];
  const float* pos    = (const float*)d_in[1];
  const int*   batch  = (const int*)  d_in[2];
  const int*   esrc   = (const int*)  d_in[3];
  const int*   edst   = (const int*)  d_in[4];
  const float* eshift = (const float*)d_in[5];
  const float* cell   = (const float*)d_in[6];
  const float* W1     = (const float*)d_in[7];
  const float* b1     = (const float*)d_in[8];
  const float* W2     = (const float*)d_in[9];
  const float* b2     = (const float*)d_in[10];
  const float* W3     = (const float*)d_in[11];
  const float* b3     = (const float*)d_in[12];

  float*    out = (float*)d_out;
  float*    cnt = (float*)d_ws;                              // 16 KB
  _Float16* pk  = (_Float16*)((char*)d_ws + 16384);          // 100 KB packed weights

  const int n = N_NODES * DIM_IN;
  const int zgrid = (n + 255) / 256;

  e3conv_zero<<<zgrid, 256, 0, stream>>>(out, cnt, n);
  e3conv_pack<<<NFRAG, 32, 0, stream>>>(W1, W2, W3, pk);
  e3conv_main<<<BLOCKS, WAVES * 32, 0, stream>>>(f_in, pos, batch, esrc, edst,
                                                 eshift, cell, b1, b2, b3, pk,
                                                 out, cnt);
  e3conv_norm<<<zgrid, 256, 0, stream>>>(out, cnt, n);
}
